// GlobalSupportSummarizer_13022340842030
// MI455X (gfx1250) — matrix operand order repaired and checked by reading, not compiled, not run on hardware
//
#include <hip/hip_runtime.h>
#include <hip/hip_bf16.h>

#define P_DIM 128
#define T_DIM 2048
#define H_DIM 1024
#define S_SLICES 8
#define T_CHUNK (T_DIM / S_SLICES)   // 256

typedef __attribute__((ext_vector_type(16))) __bf16 v16bf;
typedef __attribute__((ext_vector_type(8)))  float  v8f;

// ---------------------------------------------------------------------------
// bf16 (raw ushort) -> f32 helpers
// ---------------------------------------------------------------------------
__device__ __forceinline__ float bf16_to_f32(unsigned short u) {
    return __uint_as_float(((unsigned int)u) << 16);
}
__device__ __forceinline__ void bf16x8_to_f32(const uint4 v, float* f) {
    const unsigned short* u = reinterpret_cast<const unsigned short*>(&v);
#pragma unroll
    for (int i = 0; i < 8; ++i) f[i] = bf16_to_f32(u[i]);
}

// ---------------------------------------------------------------------------
// Phase 1: streaming reduction over T (the 1 GiB, bandwidth-bound phase).
// grid = (S_SLICES, P), block = 128 threads. Each thread owns 8 contiguous
// bf16 columns (16B vector loads, fully coalesced across the wave).
// Writes deterministic per-slice partials (sum of xs+ys, max of xs,ys).
// ---------------------------------------------------------------------------
__global__ void reduce_tokens_kernel(const unsigned short* __restrict__ xs,
                                     const unsigned short* __restrict__ ys,
                                     float* __restrict__ sumPart,
                                     float* __restrict__ maxPart) {
    const int p   = blockIdx.y;
    const int s   = blockIdx.x;
    const int tid = threadIdx.x;          // 0..127
    const int col = tid * 8;              // 8 bf16 per thread covers H=1024

    const size_t base = ((size_t)p * T_DIM + (size_t)s * T_CHUNK) * H_DIM + col;
    const unsigned short* xp = xs + base;
    const unsigned short* yp = ys + base;

    float sum[8], mx[8];
#pragma unroll
    for (int i = 0; i < 8; ++i) { sum[i] = 0.0f; mx[i] = -INFINITY; }

#pragma unroll 4
    for (int t = 0; t < T_CHUNK; ++t) {
        uint4 xv = *reinterpret_cast<const uint4*>(xp);
        uint4 yv = *reinterpret_cast<const uint4*>(yp);
        // prefetch a few rows ahead (lowers to global_prefetch_b8 on gfx1250)
        __builtin_prefetch(xp + 8 * H_DIM, 0, 0);
        __builtin_prefetch(yp + 8 * H_DIM, 0, 0);
        float xf[8], yf[8];
        bf16x8_to_f32(xv, xf);
        bf16x8_to_f32(yv, yf);
#pragma unroll
        for (int i = 0; i < 8; ++i) {
            sum[i] += xf[i] + yf[i];
            mx[i]   = fmaxf(mx[i], fmaxf(xf[i], yf[i]));
        }
        xp += H_DIM;
        yp += H_DIM;
    }

    float* sp = sumPart + (((size_t)p * S_SLICES + s) << 10) + col;
    float* mp = maxPart + (((size_t)p * S_SLICES + s) << 10) + col;
    *reinterpret_cast<float4*>(sp)     = make_float4(sum[0], sum[1], sum[2], sum[3]);
    *reinterpret_cast<float4*>(sp + 4) = make_float4(sum[4], sum[5], sum[6], sum[7]);
    *reinterpret_cast<float4*>(mp)     = make_float4(mx[0], mx[1], mx[2], mx[3]);
    *reinterpret_cast<float4*>(mp + 4) = make_float4(mx[4], mx[5], mx[6], mx[7]);
}

// ---------------------------------------------------------------------------
// Phase 2: combine slice partials -> h[P,H] in bf16 (WMMA A operand).
// h = 0.5 * ( (sum_x + sum_y) / (2T)  +  max_xy )
// ---------------------------------------------------------------------------
__global__ void combine_kernel(const float* __restrict__ sumPart,
                               const float* __restrict__ maxPart,
                               __hip_bfloat16* __restrict__ hbf) {
    const int idx = blockIdx.x * blockDim.x + threadIdx.x;  // 0..P*H-1
    const int p = idx >> 10;
    const int c = idx & 1023;
    float s = 0.0f, m = -INFINITY;
#pragma unroll
    for (int sl = 0; sl < S_SLICES; ++sl) {
        const size_t o = (((size_t)p * S_SLICES + sl) << 10) + c;
        s += sumPart[o];
        m  = fmaxf(m, maxPart[o]);
    }
    const float mean_xy = s * (0.5f / (float)T_DIM);  // (Sx+Sy)/(2T)
    hbf[idx] = __float2bfloat16(0.5f * (mean_xy + m));
}

// ---------------------------------------------------------------------------
// Phase 3: g = h @ W^T + b via v_wmma_f32_16x16x32_bf16.
// g[m,n] = sum_k h[m,k] * W[n,k]  (W row-major [n][k] => B columns are W rows,
// contiguous in k -> coalescible 16B loads, no transpose needed).
// One wave per 16x16 output tile; 8 M-tiles x 64 N-tiles = 512 tiles.
// grid = 64 blocks x 256 threads (8 waves).
// ---------------------------------------------------------------------------
__global__ void proj_wmma_kernel(const unsigned short* __restrict__ hA,   // h bf16 [128][1024]
                                 const unsigned short* __restrict__ Wb,   // W bf16 [1024][1024]
                                 const __hip_bfloat16* __restrict__ bias, // [1024]
                                 float* __restrict__ g) {                 // [128][1024] f32
    const int lane = threadIdx.x & 31;
    const int wave = threadIdx.x >> 5;
    const int tile = blockIdx.x * 8 + wave;   // 0..511
    const int m0   = (tile & 7) * 16;         // 8 M tiles
    const int n0   = (tile >> 3) * 16;        // 64 N tiles
    const int hi   = lane >> 4;               // lane group 0/1
    const int lr   = lane & 15;

    v8f acc = {0.f, 0.f, 0.f, 0.f, 0.f, 0.f, 0.f, 0.f};

    for (int k0 = 0; k0 < H_DIM; k0 += 32) {
        union { v16bf v; uint4 q[2]; } a, b;
        // A 16x32 bf16 layout: lane row m0+lr; K chunks {hi*8..+7} and {16+hi*8..+7}
        const unsigned short* ap = hA + (size_t)(m0 + lr) * H_DIM + k0 + hi * 8;
        a.q[0] = *reinterpret_cast<const uint4*>(ap);
        a.q[1] = *reinterpret_cast<const uint4*>(ap + 16);
        // B 32x16 bf16 layout: lane col n0+lr; K chunks {hi*8..+7} and {16+hi*8..+7}
        const unsigned short* bp = Wb + (size_t)(n0 + lr) * H_DIM + k0 + hi * 8;
        b.q[0] = *reinterpret_cast<const uint4*>(bp);
        b.q[1] = *reinterpret_cast<const uint4*>(bp + 16);

        acc = __builtin_amdgcn_wmma_f32_16x16x32_bf16(
            /*neg_a=*/false, a.v, /*neg_b=*/false, b.v,
            /*c_mod=*/(short)0, acc, /*reuse_a=*/false, /*reuse_b=*/false);
    }

    const float bias_f = __bfloat162float(bias[n0 + lr]);
#pragma unroll
    for (int r = 0; r < 8; ++r) {
        const int row = m0 + r + hi * 8;      // C/D layout: VGPR r -> M=r (+8 hi lanes)
        g[(size_t)row * H_DIM + n0 + lr] = acc[r] + bias_f;
    }
}

// ---------------------------------------------------------------------------
// Phase 4a: LayerNorm each row of g in parallel: one block per row (p),
// 256 threads, thread owns 4 contiguous columns. Row stats via wave32
// shuffle reductions + one 8-entry LDS combine (2 barriers/row total).
// Writes normed[P,H] fp32.
// ---------------------------------------------------------------------------
__global__ void ln_rows_kernel(const float* __restrict__ g,
                               const unsigned short* __restrict__ gamma,
                               const unsigned short* __restrict__ beta,
                               float* __restrict__ normed) {
    const int p    = blockIdx.x;
    const int tid  = threadIdx.x;     // 0..255
    const int lane = tid & 31;
    const int wv   = tid >> 5;        // 0..7

    const float4 v = reinterpret_cast<const float4*>(g + (size_t)p * H_DIM)[tid];
    float s = v.x + v.y + v.z + v.w;
    float q = v.x * v.x + v.y * v.y + v.z * v.z + v.w * v.w;

#pragma unroll
    for (int off = 16; off > 0; off >>= 1) {   // wave32 reduction
        s += __shfl_down(s, off, 32);
        q += __shfl_down(q, off, 32);
    }

    __shared__ float ws_s[8], ws_q[8];
    __shared__ float mu_sh, rstd_sh;
    if (lane == 0) { ws_s[wv] = s; ws_q[wv] = q; }
    __syncthreads();
    if (tid == 0) {
        float S = 0.0f, Q = 0.0f;
#pragma unroll
        for (int i = 0; i < 8; ++i) { S += ws_s[i]; Q += ws_q[i]; }
        const float mu  = S * (1.0f / (float)H_DIM);
        const float var = Q * (1.0f / (float)H_DIM) - mu * mu;
        mu_sh   = mu;
        rstd_sh = rsqrtf(var + 1e-5f);
    }
    __syncthreads();
    const float mu = mu_sh, r = rstd_sh;

    const ushort4 gv = reinterpret_cast<const ushort4*>(gamma)[tid];
    const ushort4 bv = reinterpret_cast<const ushort4*>(beta)[tid];
    float4 o;
    o.x = (v.x - mu) * r * bf16_to_f32(gv.x) + bf16_to_f32(bv.x);
    o.y = (v.y - mu) * r * bf16_to_f32(gv.y) + bf16_to_f32(bv.y);
    o.z = (v.z - mu) * r * bf16_to_f32(gv.z) + bf16_to_f32(bv.z);
    o.w = (v.w - mu) * r * bf16_to_f32(gv.w) + bf16_to_f32(bv.w);
    reinterpret_cast<float4*>(normed + (size_t)p * H_DIM)[tid] = o;
}

// ---------------------------------------------------------------------------
// Phase 4b: column mean over P + per-block NaN flag (deterministic slots).
// 4 blocks x 256 threads; column reads are coalesced per p-row.
// ---------------------------------------------------------------------------
__global__ void colmean_kernel(const float* __restrict__ normed,
                               float* __restrict__ gctx,
                               int* __restrict__ nanPart) {
    const int j = blockIdx.x * 256 + threadIdx.x;   // 0..1023
    float s = 0.0f;
#pragma unroll 4
    for (int p = 0; p < P_DIM; ++p) s += normed[(size_t)p * H_DIM + j];
    s *= (1.0f / (float)P_DIM);
    gctx[j] = s;

    __shared__ int f;
    if (threadIdx.x == 0) f = 0;
    __syncthreads();
    if (s != s) atomicOr(&f, 1);
    __syncthreads();
    if (threadIdx.x == 0) nanPart[blockIdx.x] = f;
}

// ---------------------------------------------------------------------------
// Phase 4c: global all-or-nothing NaN guard, emit bf16.
// ---------------------------------------------------------------------------
__global__ void finalize_kernel(const float* __restrict__ gctx,
                                const int* __restrict__ nanPart,
                                __hip_bfloat16* __restrict__ out) {
    const int j = blockIdx.x * 256 + threadIdx.x;   // 0..1023
    const int f = nanPart[0] | nanPart[1] | nanPart[2] | nanPart[3];
    out[j] = __float2bfloat16(f ? 0.0f : gctx[j]);
}

// ---------------------------------------------------------------------------
// Launch
// ---------------------------------------------------------------------------
extern "C" void kernel_launch(void* const* d_in, const int* in_sizes, int n_in,
                              void* d_out, int out_size, void* d_ws, size_t ws_size,
                              hipStream_t stream) {
    const unsigned short*  xs    = (const unsigned short*)d_in[0];
    const unsigned short*  ys    = (const unsigned short*)d_in[1];
    const unsigned short*  W     = (const unsigned short*)d_in[2];
    const __hip_bfloat16*  b     = (const __hip_bfloat16*)d_in[3];
    const unsigned short*  gamma = (const unsigned short*)d_in[4];
    const unsigned short*  beta  = (const unsigned short*)d_in[5];

    char* ws = (char*)d_ws;
    const size_t sumBytes  = (size_t)P_DIM * S_SLICES * H_DIM * sizeof(float); // 4 MB
    const size_t hBytes    = (size_t)P_DIM * H_DIM * sizeof(__hip_bfloat16);   // 256 KB
    const size_t gBytes    = (size_t)P_DIM * H_DIM * sizeof(float);            // 512 KB

    float*          sumPart = (float*)(ws);
    float*          maxPart = (float*)(ws + sumBytes);
    __hip_bfloat16* hbf     = (__hip_bfloat16*)(ws + 2 * sumBytes);
    float*          g       = (float*)(ws + 2 * sumBytes + hBytes);
    float*          normed  = (float*)(ws + 2 * sumBytes + hBytes + gBytes);
    float*          gctx    = (float*)(ws + 2 * sumBytes + hBytes + 2 * gBytes);
    int*            nanPart = (int*)(ws + 2 * sumBytes + hBytes + 2 * gBytes + H_DIM * sizeof(float));

    // Phase 1: 1 GiB streaming reduction (bandwidth bound, ~46 us at 23.3 TB/s)
    reduce_tokens_kernel<<<dim3(S_SLICES, P_DIM), 128, 0, stream>>>(xs, ys, sumPart, maxPart);
    // Phase 2: combine slice partials -> h (bf16)
    combine_kernel<<<(P_DIM * H_DIM) / 256, 256, 0, stream>>>(sumPart, maxPart, hbf);
    // Phase 3: projection GEMM via bf16 WMMA
    proj_wmma_kernel<<<64, 256, 0, stream>>>((const unsigned short*)hbf, W, b, g);
    // Phase 4a: per-row LayerNorm, 128 parallel blocks
    ln_rows_kernel<<<P_DIM, 256, 0, stream>>>(g, gamma, beta, normed);
    // Phase 4b: column mean over P + deterministic NaN partials
    colmean_kernel<<<H_DIM / 256, 256, 0, stream>>>(normed, gctx, nanPart);
    // Phase 4c: global NaN guard -> bf16 out[1024]
    finalize_kernel<<<H_DIM / 256, 256, 0, stream>>>(gctx, nanPart, (__hip_bfloat16*)d_out);
}
